// DAWN_83726092468704
// MI455X (gfx1250) — compile-verified, matrix operand order checked by reading, not run on hardware
//
#include <hip/hip_runtime.h>

// ---------------- problem constants ----------------
#define DMODEL 1024
#define SEQ    1024
#define BATCH  4
#define NHEADS 16
#define DHEAD  64
#define TOK    4096              // BATCH*SEQ
#define DSPACE 128
#define MAXK   128
#define NC     64                // clusters per gate
#define KC     8                 // clusters selected

typedef __attribute__((ext_vector_type(16))) __bf16 v16bf;
typedef __attribute__((ext_vector_type(8)))  float  v8f;
typedef int v4i_async __attribute__((vector_size(16)));
typedef __bf16 bf16;

// ---------------- small device helpers ----------------
__device__ __forceinline__ bf16 f2bf(float f) {
  union { float f; unsigned u; } v; v.f = f;
  unsigned r = v.u + 0x7fffu + ((v.u >> 16) & 1u);   // RNE
  unsigned short h = (unsigned short)(r >> 16);
  return __builtin_bit_cast(bf16, h);
}

__device__ __forceinline__ float blockReduceSum(float v, float* red, int nwaves) {
  #pragma unroll
  for (int o = 16; o > 0; o >>= 1) v += __shfl_down(v, o, 32);
  int w = threadIdx.x >> 5, lane = threadIdx.x & 31;
  __syncthreads();
  if (lane == 0) red[w] = v;
  __syncthreads();
  if (threadIdx.x == 0) { float s = 0.f; for (int i = 0; i < nwaves; ++i) s += red[i]; red[0] = s; }
  __syncthreads();
  return red[0];
}

__device__ __forceinline__ float blockReduceMax(float v, float* red, int nwaves) {
  #pragma unroll
  for (int o = 16; o > 0; o >>= 1) v = fmaxf(v, __shfl_down(v, o, 32));
  int w = threadIdx.x >> 5, lane = threadIdx.x & 31;
  __syncthreads();
  if (lane == 0) red[w] = v;
  __syncthreads();
  if (threadIdx.x == 0) { float s = red[0]; for (int i = 1; i < nwaves; ++i) s = fmaxf(s, red[i]); red[0] = s; }
  __syncthreads();
  return red[0];
}

// WMMA wrapper: D = A(16x32 bf16) * B(32x16 bf16) + C(16x16 f32)
__device__ __forceinline__ v8f wmma_bf16(v16bf a, v16bf b, v8f c) {
  return __builtin_amdgcn_wmma_f32_16x16x32_bf16(false, a, false, b, (short)0, c, false, false);
}

// A fragment: rows m0..m0+15 (row-major, leading dim ld), K slab [k0,k0+32)
// lane l: row = m0 + (l&15), half = l>>4 ; elems e<8 -> k = k0+8*half+e ; e>=8 -> k = k0+16+8*half+(e-8)
__device__ __forceinline__ v16bf load_frag_rm(const bf16* __restrict__ M, int ld, int r0, int k0) {
  int lane = threadIdx.x & 31;
  int r = r0 + (lane & 15);
  int half = lane >> 4;
  const bf16* p = M + (size_t)r * ld + k0 + half * 8;
  v16bf f;
  #pragma unroll
  for (int e = 0; e < 8; ++e) f[e] = p[e];
  p += 16;
  #pragma unroll
  for (int e = 0; e < 8; ++e) f[8 + e] = p[e];
  return f;
}

// ---------------- prep kernels ----------------
__global__ void rownorm_kernel(const float* __restrict__ src, float* __restrict__ dst,
                               bf16* __restrict__ dstbf, int cols) {
  __shared__ float red[8];
  int r = blockIdx.x;
  const float* s = src + (size_t)r * cols;
  float acc = 0.f;
  for (int c = threadIdx.x; c < cols; c += blockDim.x) { float v = s[c]; acc += v * v; }
  float n = blockReduceSum(acc, red, (blockDim.x + 31) >> 5);
  float inv = 1.0f / (sqrtf(n) + 1e-8f);
  for (int c = threadIdx.x; c < cols; c += blockDim.x) {
    float v = s[c] * inv;
    dst[(size_t)r * cols + c] = v;
    if (dstbf) dstbf[(size_t)r * cols + c] = f2bf(v);
  }
}

__global__ void transpose_bf16_kernel(const float* __restrict__ src, bf16* __restrict__ dst,
                                      int R, int C) {
  int i = blockIdx.x * 256 + threadIdx.x;
  if (i >= R * C) return;
  int r = i / C, c = i % C;
  dst[(size_t)c * R + r] = f2bf(src[i]);
}

__global__ void slice_bf16_kernel(const float* __restrict__ src, int ld, int off, int w,
                                  bf16* __restrict__ dst, int rows) {
  int i = blockIdx.x * 256 + threadIdx.x;
  if (i >= rows * w) return;
  int r = i / w, c = i % w;
  dst[i] = f2bf(src[(size_t)r * ld + off + c]);
}

__global__ void zero_kernel(float* __restrict__ p, int n) {
  int i = blockIdx.x * 256 + threadIdx.x;
  if (i < n) p[i] = 0.f;
}

__global__ void add_kernel(const float* __restrict__ a, const float* __restrict__ b,
                           float* __restrict__ c, int n) {
  int i = blockIdx.x * 256 + threadIdx.x;
  if (i < n) c[i] = a[i] + b[i];
}

// ---------------- layernorm ----------------
__global__ void layernorm_kernel(const float* __restrict__ x, const float* __restrict__ scale,
                                 const float* __restrict__ bias, float* __restrict__ out,
                                 bf16* __restrict__ outbf) {
  __shared__ float red[8];
  int t = blockIdx.x;
  const float* xr = x + (size_t)t * DMODEL;
  float s = 0.f;
  for (int d = threadIdx.x; d < DMODEL; d += 256) s += xr[d];
  float mean = blockReduceSum(s, red, 8) * (1.0f / DMODEL);
  float v = 0.f;
  for (int d = threadIdx.x; d < DMODEL; d += 256) { float dd = xr[d] - mean; v += dd * dd; }
  float var = blockReduceSum(v, red, 8) * (1.0f / DMODEL);
  float inv = rsqrtf(var + 1e-6f);
  for (int d = threadIdx.x; d < DMODEL; d += 256) {
    float o = (xr[d] - mean) * inv * scale[d] + bias[d];
    out[(size_t)t * DMODEL + d] = o;
    outbf[(size_t)t * DMODEL + d] = f2bf(o);
  }
}

// ---------------- generic WMMA GEMM with async B-tile staging ----------------
// C[M,N] = A[M,K](bf16) * Bt[N,K](bf16) + bias + addend
// grid: (N/16, ceil(M/128)), block 256 (8 waves, each wave one 16x16 tile of the
// same 16-column strip). The 16x32 B tile is staged once per block into LDS via
// the CDNA5 async global->LDS path (ASYNCcnt) and shared by all 8 waves.
__global__ void gemm_bf16_kernel(const bf16* __restrict__ A, const bf16* __restrict__ Bt,
                                 const float* __restrict__ bias, const float* __restrict__ addend,
                                 float* __restrict__ C, bf16* __restrict__ Cbf,
                                 int M, int N, int K) {
  __shared__ bf16 s_b[16 * 32];      // Bt tile: 16 n-rows x 32 k
  int tid = threadIdx.x;
  int wave = tid >> 5;
  int m0 = blockIdx.y * 128 + wave * 16;
  int n0 = blockIdx.x * 16;
  bool active = (m0 < M);            // wave-uniform
  v8f acc = {};
  for (int k0 = 0; k0 < K; k0 += 32) {
    __syncthreads();                 // protect s_b from previous iteration's readers
    if (tid < 64) {                  // waves 0..1: stage 16x32 bf16 = 1KB as 64x16B
      int row = tid >> 2, chunk = tid & 3;
      bf16* g = const_cast<bf16*>(Bt + (size_t)(n0 + row) * K + k0 + chunk * 8);
#if __has_builtin(__builtin_amdgcn_global_load_async_to_lds_b128)
      __builtin_amdgcn_global_load_async_to_lds_b128(
          (__attribute__((address_space(1))) v4i_async*)(void*)g,
          (__attribute__((address_space(3))) v4i_async*)(void*)&s_b[tid * 8],
          0, 0);
#else
      *(int4*)&s_b[tid * 8] = *(const int4*)g;
#endif
    }
#if __has_builtin(__builtin_amdgcn_s_wait_asynccnt)
    if (wave < 2) __builtin_amdgcn_s_wait_asynccnt(0);
#endif
    __syncthreads();                 // publish s_b to all waves
    if (active) {
      v16bf a = load_frag_rm(A, K, m0, k0);
      v16bf b = load_frag_rm(s_b, 32, 0, 0);   // ds_load path
      acc = wmma_bf16(a, b, acc);
    }
  }
  if (!active) return;
  int lane = tid & 31;
  int n = n0 + (lane & 15);
  int mb = m0 + ((lane >> 4) << 3);
  float bi = bias ? bias[n] : 0.f;
  #pragma unroll
  for (int r = 0; r < 8; ++r) {
    size_t idx = (size_t)(mb + r) * N + n;
    float v = acc[r] + bi;
    if (addend) v += addend[idx];
    if (C)   C[idx]   = v;
    if (Cbf) Cbf[idx] = f2bf(v);
  }
}

// ---------------- tau (tiny GEMV, N<=3) ----------------
__global__ void tau_kernel(const float* __restrict__ h, const float* __restrict__ w,
                           const float* __restrict__ b, int ntau, float* __restrict__ out) {
  __shared__ float red[8];
  int t = blockIdx.x;
  const float* hr = h + (size_t)t * DMODEL;
  for (int o = 0; o < ntau; ++o) {
    float acc = 0.f;
    for (int d = threadIdx.x; d < DMODEL; d += 256) acc += hr[d] * w[d * ntau + o];
    float s = blockReduceSum(acc, red, 8);
    if (threadIdx.x == 0) out[(size_t)t * ntau + o] = s + b[o];
  }
}

// ---------------- hierarchical gate (one token per block) ----------------
__global__ void gate_kernel(const float* __restrict__ hg, int hg_ld, int hg_off,
                            const float* __restrict__ tau, int tau_ld, int tau_off,
                            const float* __restrict__ cs,    // [T,64] cluster scores
                            const float* __restrict__ embn,  // [Nn,128] normalized neuron emb
                            int CS, int NCAND,
                            float* __restrict__ gate_vals,   // [T,NCAND]
                            int* __restrict__ cids,          // [T,8]
                            float* __restrict__ cfreq,       // [64]
                            float* __restrict__ nsum) {      // [Nn]
  __shared__ float s_cs[NC];
  __shared__ int   s_mark[NC];
  __shared__ int   s_pick[KC];
  __shared__ float s_h[DSPACE];
  __shared__ float s_eg[1024];
  __shared__ float s_sort[1024];
  __shared__ float red[8];
  int t = blockIdx.x, tid = threadIdx.x;

  if (tid < NC) { s_cs[tid] = cs[(size_t)t * NC + tid]; s_mark[tid] = 0; }
  if (tid < DSPACE) s_h[tid] = hg[(size_t)t * hg_ld + hg_off + tid];
  __syncthreads();

  if (tid == 0) {
    // cluster softmax -> frequency accumulator (for aux)
    float mx = -3.4e38f;
    for (int c = 0; c < NC; ++c) mx = fmaxf(mx, s_cs[c]);
    float sm = 0.f;
    for (int c = 0; c < NC; ++c) sm += __expf(s_cs[c] - mx);
    float invsm = 1.0f / sm;
    for (int c = 0; c < NC; ++c)
      atomicAdd(&cfreq[c], __expf(s_cs[c] - mx) * invsm * (1.0f / TOK));
    // top-8 clusters
    for (int p = 0; p < KC; ++p) {
      int best = 0; float bv = -3.4e38f;
      for (int c = 0; c < NC; ++c)
        if (!s_mark[c] && s_cs[c] > bv) { bv = s_cs[c]; best = c; }
      s_mark[best] = 1; s_pick[p] = best; cids[t * KC + p] = best;
    }
  }
  __syncthreads();

  float tv = tau[(size_t)t * tau_ld + tau_off];
  for (int j = tid; j < NCAND; j += 256) {
    int nid = s_pick[j / CS] * CS + (j % CS);
    const float* e = embn + (size_t)nid * DSPACE;
    float d = 0.f;
    #pragma unroll 8
    for (int k = 0; k < DSPACE; ++k) d += s_h[k] * e[k];
    float raw = d - tv;
    float g = raw > 0.f ? raw : 1e-8f * __expf(raw);
    float eg = __expf(g) - 1.0f;
    s_eg[j] = eg; s_sort[j] = eg;
  }
  __syncthreads();

  // bitonic sort descending to find exact MAXK-th largest
  for (int k = 2; k <= NCAND; k <<= 1) {
    for (int j = k >> 1; j > 0; j >>= 1) {
      for (int i = tid; i < NCAND; i += 256) {
        int ixj = i ^ j;
        if (ixj > i) {
          float a = s_sort[i], b = s_sort[ixj];
          bool descend = ((i & k) == 0);
          if (descend ? (a < b) : (a > b)) { s_sort[i] = b; s_sort[ixj] = a; }
        }
      }
      __syncthreads();
    }
  }
  float thr = s_sort[MAXK - 1];

  float ls = 0.f, lm = -3.4e38f;
  for (int j = tid; j < NCAND; j += 256) {
    float e = s_eg[j];
    float kept = (e >= thr) ? e : 0.f;
    s_eg[j] = kept;
    ls += kept; lm = fmaxf(lm, kept);
  }
  float S = blockReduceSum(ls, red, 8);
  float M = blockReduceMax(lm, red, 8);
  float scale = tanhf(M) / (S + 1e-8f);
  for (int j = tid; j < NCAND; j += 256) {
    float g = s_eg[j] * scale;
    gate_vals[(size_t)t * NCAND + j] = g;
    if (g != 0.f) atomicAdd(&nsum[s_pick[j / CS] * CS + (j % CS)], g);
  }
}

// ---------------- sparse sense_emit (one token per block) ----------------
__global__ void sense_emit_kernel(const float* __restrict__ h,        // [T,1024]
                                  const float* __restrict__ neurons,  // [Nn,1024]
                                  const float* __restrict__ gate_vals,// [T,NCAND]
                                  const int* __restrict__ cids,       // [T,8]
                                  int CS, int NCAND,
                                  float* __restrict__ out,            // [T,1024]
                                  bf16* __restrict__ outbf) {
  __shared__ float s_h[DMODEL];
  __shared__ float s_w[1024];
  __shared__ int   s_nid[1024];
  int t = blockIdx.x, tid = threadIdx.x;
  for (int d = tid; d < DMODEL; d += 256) s_h[d] = h[(size_t)t * DMODEL + d];
  __syncthreads();
  for (int j = tid; j < NCAND; j += 256) {
    float g = gate_vals[(size_t)t * NCAND + j];
    int nid = cids[t * KC + j / CS] * CS + (j % CS);
    s_nid[j] = nid;
    float w = 0.f;
    if (g != 0.f) {
      const float* nr = neurons + (size_t)nid * DMODEL;
      float d = 0.f;
      #pragma unroll 4
      for (int k = 0; k < DMODEL; ++k) d += s_h[k] * nr[k];
      w = d * g;
    }
    s_w[j] = w;
  }
  __syncthreads();
  for (int d = tid; d < DMODEL; d += 256) {
    float acc = 0.f;
    for (int j = 0; j < NCAND; ++j) {
      float w = s_w[j];                         // uniform across lanes
      if (w != 0.f) {
        const float* nr = neurons + (size_t)s_nid[j] * DMODEL;
        __builtin_prefetch(nr + d + 256, 0, 1);
        acc += w * nr[d];
      }
    }
    float o = acc;
    out[(size_t)t * DMODEL + d] = o;
    if (outbf) outbf[(size_t)t * DMODEL + d] = f2bf(o);
  }
}

// ---------------- attention: one (b,h,16-query tile) per block ----------------
// dyn LDS: 16*1024 f32 scores + 16 f32 scratch
__global__ void attn_kernel(const bf16* __restrict__ Qbf, const bf16* __restrict__ Kbf,
                            const bf16* __restrict__ Vbf, bf16* __restrict__ Obf) {
  extern __shared__ float smem[];
  float* sc  = smem;                 // [16][1024]
  float* red = smem + 16 * SEQ;      // [8]
  int qb = blockIdx.x, bh = blockIdx.y;
  int b = bh / NHEADS, h = bh % NHEADS;
  int q0 = qb * 16;
  const bf16* Qh = Qbf + (size_t)b * SEQ * DMODEL + h * DHEAD;
  const bf16* Kh = Kbf + (size_t)b * SEQ * DMODEL + h * DHEAD;
  const bf16* Vh = Vbf + (size_t)b * SEQ * DMODEL + h * DHEAD;
  int wave = threadIdx.x >> 5, lane = threadIdx.x & 31;
  const float scale = 0.125f;        // 1/sqrt(64)

  // phase 1: scores = Q * K^T (per wave: strided key tiles)
  v16bf aq0 = load_frag_rm(Qh, DMODEL, q0, 0);
  v16bf aq1 = load_frag_rm(Qh, DMODEL, q0, 32);
  for (int kt = wave; kt < SEQ / 16; kt += 8) {
    v8f acc = {};
    v16bf b0 = load_frag_rm(Kh, DMODEL, kt * 16, 0);
    v16bf b1 = load_frag_rm(Kh, DMODEL, kt * 16, 32);
    acc = wmma_bf16(aq0, b0, acc);
    acc = wmma_bf16(aq1, b1, acc);
    int n = kt * 16 + (lane & 15);
    int mb = (lane >> 4) << 3;
    #pragma unroll
    for (int r = 0; r < 8; ++r) {
      int m = mb + r;
      float v = acc[r] * scale;
      if (n > q0 + m) v = -3.4e38f;  // causal mask
      sc[m * SEQ + n] = v;
    }
  }
  __syncthreads();

  // phase 2: row-wise softmax over 1024 keys
  for (int m = 0; m < 16; ++m) {
    float mx = -3.4e38f;
    for (int j = threadIdx.x; j < SEQ; j += 256) mx = fmaxf(mx, sc[m * SEQ + j]);
    mx = blockReduceMax(mx, red, 8);
    float sum = 0.f;
    for (int j = threadIdx.x; j < SEQ; j += 256) {
      float e = __expf(sc[m * SEQ + j] - mx);
      sc[m * SEQ + j] = e;
      sum += e;
    }
    sum = blockReduceSum(sum, red, 8);
    float inv = 1.0f / sum;
    for (int j = threadIdx.x; j < SEQ; j += 256) sc[m * SEQ + j] *= inv;
    __syncthreads();
  }
  __syncthreads();

  // phase 3: out = P * V  (4 waves, one 16-dim tile each, K = 1024 keys)
  if (wave < 4) {
    int n0 = wave * 16;
    v8f acc = {};
    int m = lane & 15, half = lane >> 4;
    int nn = n0 + (lane & 15);
    for (int k0 = 0; k0 < SEQ; k0 += 32) {
      v16bf a, bb;
      const float* p = sc + m * SEQ + k0 + half * 8;
      #pragma unroll
      for (int e = 0; e < 8; ++e) a[e] = f2bf(p[e]);
      p += 16;
      #pragma unroll
      for (int e = 0; e < 8; ++e) a[8 + e] = f2bf(p[e]);
      int kb0 = k0 + half * 8;
      #pragma unroll
      for (int e = 0; e < 8; ++e) bb[e] = Vh[(size_t)(kb0 + e) * DMODEL + nn];
      int kb1 = k0 + 16 + half * 8;
      #pragma unroll
      for (int e = 0; e < 8; ++e) bb[8 + e] = Vh[(size_t)(kb1 + e) * DMODEL + nn];
      acc = wmma_bf16(a, bb, acc);
    }
    int mb = (lane >> 4) << 3;
    #pragma unroll
    for (int r = 0; r < 8; ++r) {
      int q = q0 + mb + r;
      Obf[((size_t)b * SEQ + q) * DMODEL + h * DHEAD + nn] = f2bf(acc[r]);
    }
  }
}

// ---------------- aux losses ----------------
__global__ void aux_kernel(const float* __restrict__ cfreq, const float* __restrict__ nsum,
                           float* __restrict__ out) {
  __shared__ float red[8];
  float acc = 0.f;
  for (int i = threadIdx.x; i < 4 * NC; i += 256) {          // 4 gates x 64 clusters
    float d = cfreq[i] - (1.0f / NC);
    acc += d * d * NC;
  }
  const float invT = 1.0f / TOK;
  for (int i = threadIdx.x; i < 3 * 4096; i += 256) {        // Q,K,V neuron aux (N=4096)
    float d = nsum[i] * invT - (1.0f / 4096.0f);
    acc += d * d * 4096.0f;
  }
  for (int i = threadIdx.x; i < 8192; i += 256) {            // know neuron aux (N=8192)
    float d = nsum[3 * 4096 + i] * invT - (1.0f / 8192.0f);
    acc += d * d * 8192.0f;
  }
  float s = blockReduceSum(acc, red, 8);
  if (threadIdx.x == 0) out[0] = s;
}

// ---------------- workspace layout ----------------
static constexpr size_t A256(size_t x) { return (x + 255) & ~size_t(255); }
static constexpr size_t OFF_EMBN  = 0;                                          // 16384x128 f32
static constexpr size_t OFF_CEN   = A256(OFF_EMBN  + 16384ull * 128 * 4);       // 192x128 f32
static constexpr size_t OFF_CENBF = A256(OFF_CEN   + 192ull * 128 * 4);         // 192x128 bf16
static constexpr size_t OFF_PAT   = A256(OFF_CENBF + 192ull * 128 * 2);         // 384x1024 bf16
static constexpr size_t OFF_PKT   = A256(OFF_PAT   + 384ull * 1024 * 2);        // 128x1024 bf16
static constexpr size_t OFF_EOT   = A256(OFF_PKT   + 128ull * 1024 * 2);        // 1024x1024 bf16
static constexpr size_t OFF_H1    = A256(OFF_EOT   + 1024ull * 1024 * 2);
static constexpr size_t OFF_H1BF  = A256(OFF_H1    + (size_t)TOK * 1024 * 4);
static constexpr size_t OFF_HALL  = A256(OFF_H1BF  + (size_t)TOK * 1024 * 2);   // T x 384
static constexpr size_t OFF_TAU   = A256(OFF_HALL  + (size_t)TOK * 384 * 4);    // T x 3
static constexpr size_t OFF_HQBF  = A256(OFF_TAU   + (size_t)TOK * 3 * 4);
static constexpr size_t OFF_HKBF  = A256(OFF_HQBF  + (size_t)TOK * 128 * 2);
static constexpr size_t OFF_HVBF  = A256(OFF_HKBF  + (size_t)TOK * 128 * 2);
static constexpr size_t OFF_CSQ   = A256(OFF_HVBF  + (size_t)TOK * 128 * 2);
static constexpr size_t OFF_CSK   = A256(OFF_CSQ   + (size_t)TOK * 64 * 4);
static constexpr size_t OFF_CSV   = A256(OFF_CSK   + (size_t)TOK * 64 * 4);
static constexpr size_t OFF_CSN   = A256(OFF_CSV   + (size_t)TOK * 64 * 4);
static constexpr size_t OFF_CIDQ  = A256(OFF_CSN   + (size_t)TOK * 64 * 4);
static constexpr size_t OFF_CIDK  = A256(OFF_CIDQ  + (size_t)TOK * 8 * 4);
static constexpr size_t OFF_CIDV  = A256(OFF_CIDK  + (size_t)TOK * 8 * 4);
static constexpr size_t OFF_CIDN  = A256(OFF_CIDV  + (size_t)TOK * 8 * 4);
static constexpr size_t OFF_GVQ   = A256(OFF_CIDN  + (size_t)TOK * 8 * 4);
static constexpr size_t OFF_GVK   = A256(OFF_GVQ   + (size_t)TOK * 512 * 4);
static constexpr size_t OFF_GVV   = A256(OFF_GVK   + (size_t)TOK * 512 * 4);
static constexpr size_t OFF_GVN   = A256(OFF_GVV   + (size_t)TOK * 512 * 4);
static constexpr size_t OFF_QF    = A256(OFF_GVN   + (size_t)TOK * 1024 * 4);
static constexpr size_t OFF_KF    = A256(OFF_QF    + (size_t)TOK * 1024 * 4);
static constexpr size_t OFF_VF    = A256(OFF_KF    + (size_t)TOK * 1024 * 4);
static constexpr size_t OFF_QBF   = A256(OFF_VF    + (size_t)TOK * 1024 * 4);
static constexpr size_t OFF_KBF   = A256(OFF_QBF   + (size_t)TOK * 1024 * 2);
static constexpr size_t OFF_VBF   = A256(OFF_KBF   + (size_t)TOK * 1024 * 2);
static constexpr size_t OFF_OBF   = A256(OFF_VBF   + (size_t)TOK * 1024 * 2);
static constexpr size_t OFF_X2    = A256(OFF_OBF   + (size_t)TOK * 1024 * 2);
static constexpr size_t OFF_H2    = A256(OFF_X2    + (size_t)TOK * 1024 * 4);
static constexpr size_t OFF_H2BF  = A256(OFF_H2    + (size_t)TOK * 1024 * 4);
static constexpr size_t OFF_HK2   = A256(OFF_H2BF  + (size_t)TOK * 1024 * 2);
static constexpr size_t OFF_HK2BF = A256(OFF_HK2   + (size_t)TOK * 128 * 4);
static constexpr size_t OFF_TAUK  = A256(OFF_HK2BF + (size_t)TOK * 128 * 2);
static constexpr size_t OFF_KO    = A256(OFF_TAUK  + (size_t)TOK * 4);
static constexpr size_t OFF_CFREQ = A256(OFF_KO    + (size_t)TOK * 1024 * 4);   // 4x64
static constexpr size_t OFF_NSUM  = OFF_CFREQ + 256 * 4;                        // 20480 (contiguous)

extern "C" void kernel_launch(void* const* d_in, const int* in_sizes, int n_in,
                              void* d_out, int out_size, void* d_ws, size_t ws_size,
                              hipStream_t stream) {
  (void)in_sizes; (void)n_in; (void)out_size; (void)ws_size;
  const float* x          = (const float*)d_in[0];
  const float* neuron_emb = (const float*)d_in[1];
  const float* proj_attn_k= (const float*)d_in[2];
  const float* proj_attn_b= (const float*)d_in[3];
  const float* tau_attn_k = (const float*)d_in[4];
  const float* tau_attn_b = (const float*)d_in[5];
  const float* proj_know_k= (const float*)d_in[6];
  const float* proj_know_b= (const float*)d_in[7];
  const float* tau_know_k = (const float*)d_in[8];
  const float* tau_know_b = (const float*)d_in[9];
  const float* ce_qk      = (const float*)d_in[10];
  const float* ce_v       = (const float*)d_in[11];
  const float* ce_know    = (const float*)d_in[12];
  const float* qk_neurons = (const float*)d_in[13];
  const float* v_neurons  = (const float*)d_in[14];
  const float* know_neurons=(const float*)d_in[15];
  const float* expand_O   = (const float*)d_in[16];
  const float* ln1_s = (const float*)d_in[17];
  const float* ln1_b = (const float*)d_in[18];
  const float* ln2_s = (const float*)d_in[19];
  const float* ln2_b = (const float*)d_in[20];

  char* ws = (char*)d_ws;
  float* embn   = (float*)(ws + OFF_EMBN);
  float* cen    = (float*)(ws + OFF_CEN);
  bf16*  cenbf  = (bf16*) (ws + OFF_CENBF);
  bf16*  paT    = (bf16*) (ws + OFF_PAT);
  bf16*  pkT    = (bf16*) (ws + OFF_PKT);
  bf16*  eoT    = (bf16*) (ws + OFF_EOT);
  float* h1     = (float*)(ws + OFF_H1);
  bf16*  h1bf   = (bf16*) (ws + OFF_H1BF);
  float* hall   = (float*)(ws + OFF_HALL);
  float* tauA   = (float*)(ws + OFF_TAU);
  bf16*  hqbf   = (bf16*) (ws + OFF_HQBF);
  bf16*  hkbf   = (bf16*) (ws + OFF_HKBF);
  bf16*  hvbf   = (bf16*) (ws + OFF_HVBF);
  float* csq    = (float*)(ws + OFF_CSQ);
  float* csk    = (float*)(ws + OFF_CSK);
  float* csv    = (float*)(ws + OFF_CSV);
  float* csn    = (float*)(ws + OFF_CSN);
  int*   cidq   = (int*)  (ws + OFF_CIDQ);
  int*   cidk   = (int*)  (ws + OFF_CIDK);
  int*   cidv   = (int*)  (ws + OFF_CIDV);
  int*   cidn   = (int*)  (ws + OFF_CIDN);
  float* gvq    = (float*)(ws + OFF_GVQ);
  float* gvk    = (float*)(ws + OFF_GVK);
  float* gvv    = (float*)(ws + OFF_GVV);
  float* gvn    = (float*)(ws + OFF_GVN);
  float* Qf     = (float*)(ws + OFF_QF);
  float* Kf     = (float*)(ws + OFF_KF);
  float* Vf     = (float*)(ws + OFF_VF);
  bf16*  Qbf    = (bf16*) (ws + OFF_QBF);
  bf16*  Kbf    = (bf16*) (ws + OFF_KBF);
  bf16*  Vbf    = (bf16*) (ws + OFF_VBF);
  bf16*  Obf    = (bf16*) (ws + OFF_OBF);
  float* x2     = (float*)(ws + OFF_X2);
  float* h2     = (float*)(ws + OFF_H2);
  bf16*  h2bf   = (bf16*) (ws + OFF_H2BF);
  float* hk2    = (float*)(ws + OFF_HK2);
  bf16*  hk2bf  = (bf16*) (ws + OFF_HK2BF);
  float* tauK   = (float*)(ws + OFF_TAUK);
  float* ko     = (float*)(ws + OFF_KO);
  float* cfreq  = (float*)(ws + OFF_CFREQ);
  float* nsum   = (float*)(ws + OFF_NSUM);
  float* out_x  = (float*)d_out;
  float* out_aux= (float*)d_out + (size_t)TOK * DMODEL;

  const int NT = 4 * 64 + 3 * 4096 + 8192;  // cfreq + nsum element count

  // ---- prep ----
  zero_kernel<<<(NT + 255) / 256, 256, 0, stream>>>(cfreq, NT);
  rownorm_kernel<<<16384, 128, 0, stream>>>(neuron_emb, embn, (bf16*)nullptr, DSPACE);
  rownorm_kernel<<<64, 128, 0, stream>>>(ce_qk,   cen,             cenbf,             DSPACE);
  rownorm_kernel<<<64, 128, 0, stream>>>(ce_v,    cen + 64 * 128,  cenbf + 64 * 128,  DSPACE);
  rownorm_kernel<<<64, 128, 0, stream>>>(ce_know, cen + 128 * 128, cenbf + 128 * 128, DSPACE);
  transpose_bf16_kernel<<<(1024 * 384 + 255) / 256, 256, 0, stream>>>(proj_attn_k, paT, 1024, 384);
  transpose_bf16_kernel<<<(1024 * 128 + 255) / 256, 256, 0, stream>>>(proj_know_k, pkT, 1024, 128);
  transpose_bf16_kernel<<<(1024 * 1024 + 255) / 256, 256, 0, stream>>>(expand_O, eoT, 1024, 1024);

  // ---- first half ----
  layernorm_kernel<<<TOK, 256, 0, stream>>>(x, ln1_s, ln1_b, h1, h1bf);
  gemm_bf16_kernel<<<dim3(384 / 16, TOK / 128), 256, 0, stream>>>(
      h1bf, paT, proj_attn_b, nullptr, hall, nullptr, TOK, 384, 1024);
  tau_kernel<<<TOK, 256, 0, stream>>>(h1, tau_attn_k, tau_attn_b, 3, tauA);
  slice_bf16_kernel<<<(TOK * 128 + 255) / 256, 256, 0, stream>>>(hall, 384, 0,   128, hqbf, TOK);
  slice_bf16_kernel<<<(TOK * 128 + 255) / 256, 256, 0, stream>>>(hall, 384, 128, 128, hkbf, TOK);
  slice_bf16_kernel<<<(TOK * 128 + 255) / 256, 256, 0, stream>>>(hall, 384, 256, 128, hvbf, TOK);
  gemm_bf16_kernel<<<dim3(64 / 16, TOK / 128), 256, 0, stream>>>(
      hqbf, cenbf, nullptr, nullptr, csq, nullptr, TOK, 64, 128);
  gemm_bf16_kernel<<<dim3(64 / 16, TOK / 128), 256, 0, stream>>>(
      hkbf, cenbf, nullptr, nullptr, csk, nullptr, TOK, 64, 128);
  gemm_bf16_kernel<<<dim3(64 / 16, TOK / 128), 256, 0, stream>>>(
      hvbf, cenbf + 64 * 128, nullptr, nullptr, csv, nullptr, TOK, 64, 128);

  gate_kernel<<<TOK, 256, 0, stream>>>(hall, 384, 0,   tauA, 3, 0, csq, embn,               64, 512, gvq, cidq, cfreq,        nsum);
  gate_kernel<<<TOK, 256, 0, stream>>>(hall, 384, 128, tauA, 3, 1, csk, embn,               64, 512, gvk, cidk, cfreq + 64,   nsum + 4096);
  gate_kernel<<<TOK, 256, 0, stream>>>(hall, 384, 256, tauA, 3, 2, csv, embn + 4096 * 128,  64, 512, gvv, cidv, cfreq + 128,  nsum + 8192);

  sense_emit_kernel<<<TOK, 256, 0, stream>>>(h1, qk_neurons, gvq, cidq, 64, 512, Qf, Qbf);
  sense_emit_kernel<<<TOK, 256, 0, stream>>>(h1, qk_neurons, gvk, cidk, 64, 512, Kf, Kbf);
  sense_emit_kernel<<<TOK, 256, 0, stream>>>(h1, v_neurons,  gvv, cidv, 64, 512, Vf, Vbf);

  attn_kernel<<<dim3(SEQ / 16, BATCH * NHEADS), 256, (16 * SEQ + 16) * sizeof(float), stream>>>(
      Qbf, Kbf, Vbf, Obf);
  // x2 = x + attn_out @ expand_O  (residual fused as addend)
  gemm_bf16_kernel<<<dim3(1024 / 16, TOK / 128), 256, 0, stream>>>(
      Obf, eoT, nullptr, x, x2, nullptr, TOK, 1024, 1024);

  // ---- second half ----
  layernorm_kernel<<<TOK, 256, 0, stream>>>(x2, ln2_s, ln2_b, h2, h2bf);
  gemm_bf16_kernel<<<dim3(128 / 16, TOK / 128), 256, 0, stream>>>(
      h2bf, pkT, proj_know_b, nullptr, hk2, hk2bf, TOK, 128, 1024);
  tau_kernel<<<TOK, 256, 0, stream>>>(h2, tau_know_k, tau_know_b, 1, tauK);
  gemm_bf16_kernel<<<dim3(64 / 16, TOK / 128), 256, 0, stream>>>(
      hk2bf, cenbf + 128 * 128, nullptr, nullptr, csn, nullptr, TOK, 64, 128);
  gate_kernel<<<TOK, 256, 0, stream>>>(hk2, 128, 0, tauK, 1, 0, csn, embn + 8192 * 128,
                                       128, 1024, gvn, cidn, cfreq + 192, nsum + 12288);
  sense_emit_kernel<<<TOK, 256, 0, stream>>>(h2, know_neurons, gvn, cidn, 128, 1024, ko, (bf16*)nullptr);

  // ---- final output + aux ----
  add_kernel<<<(TOK * DMODEL + 255) / 256, 256, 0, stream>>>(x2, ko, out_x, TOK * DMODEL);
  aux_kernel<<<1, 256, 0, stream>>>(cfreq, nsum, out_aux);
}
